// Router_17892833755767
// MI455X (gfx1250) — compile-verified, matrix operand order checked by reading, not run on hardware
//
#include <hip/hip_runtime.h>
#include <hip/hip_bf16.h>
#include <math.h>

// ---- CDNA5 WMMA types -------------------------------------------------------
typedef __attribute__((ext_vector_type(16))) __bf16 v16bf;
typedef __attribute__((ext_vector_type(8)))  float  v8f;
typedef __attribute__((ext_vector_type(4)))  float  f32x4;

#define HIDDEN            4096
#define NUM_EXPERTS       256
#define TOP_K             8
#define TOKENS_PER_BLOCK  32      // two 16-row M-tiles per wave
#define KCHUNK            32

static __device__ __forceinline__ v16bf cvt_f32x16_to_bf16(f32x4 p0, f32x4 p1,
                                                           f32x4 p2, f32x4 p3) {
  v16bf r;
#pragma unroll
  for (int i = 0; i < 4; ++i) {
    r[i]      = (__bf16)p0[i];
    r[i + 4]  = (__bf16)p1[i];
    r[i + 8]  = (__bf16)p2[i];
    r[i + 12] = (__bf16)p3[i];
  }
  return r;
}

// 16 tokens/experts x 32 K slice, fp32 -> bf16, matching the WMMA 16-bit
// operand layout (lanes 0-15: K=0-7,16-23; lanes 16-31: K=8-15,24-31).
static __device__ __forceinline__ v16bf load_tile_bf16(const float* __restrict__ row,
                                                       int kb) {
  f32x4 p0 = *(const f32x4*)(row + kb);
  f32x4 p1 = *(const f32x4*)(row + kb + 4);
  f32x4 p2 = *(const f32x4*)(row + kb + 16);
  f32x4 p3 = *(const f32x4*)(row + kb + 20);
  return cvt_f32x16_to_bf16(p0, p1, p2, p3);
}

static __device__ __forceinline__ float fast_sigmoid(float z) {
  // v_exp_f32 + v_rcp_f32 (avoid the IEEE div_scale sequence).
  return __builtin_amdgcn_rcpf(1.0f + __expf(-z));
}

__global__ __launch_bounds__(256)
void router_topk_kernel(const float* __restrict__ x,
                        const float* __restrict__ gate_w,
                        const float* __restrict__ expert_bias,
                        float* __restrict__ out_idx,
                        float* __restrict__ out_w) {
  __shared__ float s_scores[TOKENS_PER_BLOCK][NUM_EXPERTS + 8];
  __shared__ float s_bias[NUM_EXPERTS];

  const int tid  = threadIdx.x;
  const int wave = tid >> 5;          // 0..7 -> 32-expert slice
  const int lane = tid & 31;
  const int half = lane >> 4;         // K-half select (WMMA 16-bit layout)
  const int l16  = lane & 15;         // M row (A) / N column (B)

  if (tid < NUM_EXPERTS) s_bias[tid] = expert_bias[tid];

  const int tok0 = blockIdx.x * TOKENS_PER_BLOCK;

  // A: two 16-token M-tiles per wave (shared by all waves via WGP$).
  const float* __restrict__ aRow0 = x + (size_t)(tok0 + l16) * HIDDEN;
  const float* __restrict__ aRow1 = aRow0 + (size_t)16 * HIDDEN;
  // B: two 16-expert N-tiles per wave (gate_w is L2-resident: 4 MB vs 192 MB L2).
  const float* __restrict__ b0Row = gate_w + (size_t)(wave * 32 + l16) * HIDDEN;
  const float* __restrict__ b1Row = b0Row + (size_t)16 * HIDDEN;

  v8f acc00 = {}, acc01 = {}, acc10 = {}, acc11 = {};

  for (int kt = 0; kt < HIDDEN; kt += KCHUNK) {
    const int kb = kt + 8 * half;

    // Prefetch next chunk of the streaming operand.
    if (kt + KCHUNK < HIDDEN) {
      __builtin_prefetch(aRow0 + kb + KCHUNK, 0, 1);
      __builtin_prefetch(aRow1 + kb + KCHUNK, 0, 1);
    }

    v16bf A0 = load_tile_bf16(aRow0, kb);
    v16bf A1 = load_tile_bf16(aRow1, kb);
    v16bf B0 = load_tile_bf16(b0Row, kb);
    v16bf B1 = load_tile_bf16(b1Row, kb);

    // 2x2 tile grid: each operand feeds two WMMAs.
    acc00 = __builtin_amdgcn_wmma_f32_16x16x32_bf16(
        false, A0, false, B0, (short)0, acc00, false, false);
    acc01 = __builtin_amdgcn_wmma_f32_16x16x32_bf16(
        false, A0, false, B1, (short)0, acc01, false, false);
    acc10 = __builtin_amdgcn_wmma_f32_16x16x32_bf16(
        false, A1, false, B0, (short)0, acc10, false, false);
    acc11 = __builtin_amdgcn_wmma_f32_16x16x32_bf16(
        false, A1, false, B1, (short)0, acc11, false, false);
  }

  // Sigmoid in-register, park scores in LDS.
  // C/D layout: VGPR r holds M = r + 8*half, N = l16 (+tile offsets).
#pragma unroll
  for (int r = 0; r < 8; ++r) {
    const int m = r + 8 * half;
    const int n0 = wave * 32 + l16;
    s_scores[m][n0]            = fast_sigmoid(acc00[r]);
    s_scores[m][n0 + 16]       = fast_sigmoid(acc01[r]);
    s_scores[m + 16][n0]       = fast_sigmoid(acc10[r]);
    s_scores[m + 16][n0 + 16]  = fast_sigmoid(acc11[r]);
  }
  __syncthreads();

  // Biased top-8 per token; one thread per token (cost negligible vs GEMM).
  if (tid < TOKENS_PER_BLOCK) {
    const int t = tid;
    unsigned sel[NUM_EXPERTS / 32];
#pragma unroll
    for (int i = 0; i < NUM_EXPERTS / 32; ++i) sel[i] = 0u;

    int   top_i[TOP_K];
    float top_s[TOP_K];
    float ssum = 0.0f;

#pragma unroll
    for (int j = 0; j < TOP_K; ++j) {
      float best = -INFINITY;
      int   be   = 0;
      for (int e = 0; e < NUM_EXPERTS; ++e) {
        if ((sel[e >> 5] >> (e & 31)) & 1u) continue;
        const float v = s_scores[t][e] + s_bias[e];
        if (v > best) { best = v; be = e; }   // strict > keeps lowest index on ties
      }
      sel[be >> 5] |= 1u << (be & 31);
      const float sc = s_scores[t][be];
      top_i[j] = be;
      top_s[j] = sc;
      ssum += sc;
    }

    const float inv = 1.0f / ssum;
    const size_t base = (size_t)(tok0 + t) * TOP_K;
#pragma unroll
    for (int j = 0; j < TOP_K; ++j) {
      out_idx[base + j] = (float)top_i[j];
      out_w[base + j]   = top_s[j] * inv;
    }
  }
}

extern "C" void kernel_launch(void* const* d_in, const int* in_sizes, int n_in,
                              void* d_out, int out_size, void* d_ws, size_t ws_size,
                              hipStream_t stream) {
  const float* x    = (const float*)d_in[0];  // [4, 4096, 4096] fp32
  const float* gw   = (const float*)d_in[1];  // [256, 4096]     fp32
  const float* bias = (const float*)d_in[2];  // [256]           fp32

  const int total_tokens = 4 * 4096;          // 16384
  float* out      = (float*)d_out;
  float* out_idx  = out;                                  // [16384, 8] indices
  float* out_wgt  = out + (size_t)total_tokens * TOP_K;   // [16384, 8] weights

  router_topk_kernel<<<total_tokens / TOKENS_PER_BLOCK, 256, 0, stream>>>(
      x, gw, bias, out_idx, out_wgt);
}
